// MultiHeadAttentionLayer_15822659518994
// MI455X (gfx1250) — matrix operand order repaired and checked by reading, not compiled, not run on hardware
//
#include <hip/hip_runtime.h>
#include <hip/hip_bf16.h>

// ---------------------------------------------------------------------------
// MultiHeadAttentionLayer for MI455X (gfx1250, wave32, WMMA + async-to-LDS)
// B=4, S=2048, E=1024, H=16, D=64  -> M = B*S = 8192, HD = H*D = 1024
//
// Pipeline (all matmuls on v_wmma_f32_16x16x32_f16, f32 accumulate):
//   1. layernorm_f16   : x (f32) -> xn (f16)
//   2. convert_transpose (x4): W (f32 [K][N]) -> WT (f16 [N][K])
//   3. gemm128 (x3)    : Q/K/V = xn @ W + b   (Q pre-scaled by 1/sqrt(D))
//                        double-buffered LDS tiles fed by
//                        global_load_async_to_lds_b128 (ASYNCcnt)
//   4. attn_flash      : causal online-softmax attention per (b,h,q-tile)
//   5. gemm128<FINAL>  : out = attn @ Wo + bo + x   (f32 out)
//
// Workspace layout (bytes), total ~88 MB:
//   xnH @0 (16MB) | WqT @16M | WkT @18M | WvT @20M | WoT @22M (2MB each)
//   Qh @24M | Kh @40M | Vh @56M | Ah @72M  (16MB each, f16)
// ---------------------------------------------------------------------------

#define EDIM 1024
#define HEADS 16
#define DHEAD 64
#define BATCH 4
#define SEQ 2048
#define MROWS (BATCH * SEQ)   // 8192
#define HD 1024               // HEADS*DHEAD

typedef _Float16 v16h __attribute__((ext_vector_type(16)));
typedef _Float16 v8h  __attribute__((ext_vector_type(8)));
typedef float    v8f  __attribute__((ext_vector_type(8)));

// gfx1250 async copy global->LDS (ASYNCcnt-tracked), guarded so older
// toolchains (and the host pass) fall back to synchronous LDS staging.
#if defined(__gfx1250__) &&                                         \
    __has_builtin(__builtin_amdgcn_global_load_async_to_lds_b128) && \
    __has_builtin(__builtin_amdgcn_s_wait_asynccnt)
#define USE_ASYNC_COPY 1
#else
#define USE_ASYNC_COPY 0
#endif

#if USE_ASYNC_COPY
// Builtin signature (from hipcc diagnostic): param0 is int __vector(4) in
// address space 1 (global), param1 the LDS-side pointer, then imm offset/cpol.
typedef int v4i_ __attribute__((vector_size(16)));
typedef __attribute__((address_space(1))) v4i_ gv4i_t;
typedef __attribute__((address_space(3))) v4i_ lv4i_t;
static __device__ inline void async_b128(const _Float16* g, _Float16* l) {
  __builtin_amdgcn_global_load_async_to_lds_b128((gv4i_t*)g, (lv4i_t*)l, 0, 0);
}
static __device__ inline void async_wait0() {
  __builtin_amdgcn_s_wait_asynccnt(0);
}
#else
static __device__ inline void async_wait0() {}
#endif

static __device__ inline v8f v8f_zero() {
  v8f z;
#pragma unroll
  for (int i = 0; i < 8; ++i) z[i] = 0.0f;
  return z;
}

// WMMA: D = A(16x32 f16) * B(32x16 f16) + C(16x16 f32)
static __device__ inline v8f wmma_f16(v16h a, v16h b, v8f c) {
  return __builtin_amdgcn_wmma_f32_16x16x32_f16(false, a, false, b, (short)0, c,
                                                false, false);
}

// A-operand fragment (16x32, row-major source, stride in halfs).
// ISA layout: lane<16 -> row=m0+lane, elems {k0..k0+7, k0+16..k0+23}
//             lane>=16 -> row=m0+lane-16, elems {k0+8..k0+15, k0+24..k0+31}
static __device__ inline v16h load_frag_a(const _Float16* p, int row0, int k0,
                                          int stride) {
  const int lane = threadIdx.x & 31;
  const int hsel = (lane >> 4) * 8;
  const _Float16* base = p + (row0 + (lane & 15)) * stride + k0 + hsel;
  v8h lo = *(const v8h*)(base);
  v8h hi = *(const v8h*)(base + 16);
  v16h f;
#pragma unroll
  for (int i = 0; i < 8; ++i) { f[i] = lo[i]; f[i + 8] = hi[i]; }
  return f;
}

// B-operand fragment (32x16) read from *transposed* storage BT[n][k].
// ISA layout: lanes 0-15 hold K={k0..k0+7, k0+16..k0+23} of col n0+lane,
//             lanes 16-31 hold K={k0+8..k0+15, k0+24..k0+31} of col n0+lane-16.
static __device__ inline v16h load_frag_b(const _Float16* p, int n0, int k0,
                                          int stride) {
  const int lane = threadIdx.x & 31;
  const _Float16* base = p + (n0 + (lane & 15)) * stride + k0 + (lane >> 4) * 8;
  v8h lo = *(const v8h*)(base);
  v8h hi = *(const v8h*)(base + 16);
  v16h f;
#pragma unroll
  for (int i = 0; i < 8; ++i) { f[i] = lo[i]; f[i + 8] = hi[i]; }
  return f;
}

// ---------------------------------------------------------------------------
// 1. LayerNorm (f32 in -> f16 out), one block per row, 256 threads
// ---------------------------------------------------------------------------
__global__ __launch_bounds__(256) void layernorm_f16(
    const float* __restrict__ x, const float* __restrict__ gamma,
    const float* __restrict__ beta, _Float16* __restrict__ xn) {
  const int row = blockIdx.x;
  const int t = threadIdx.x;
  const float* xr = x + (size_t)row * EDIM;
  float4 v = ((const float4*)xr)[t];
  float s = v.x + v.y + v.z + v.w;
  float sq = v.x * v.x + v.y * v.y + v.z * v.z + v.w * v.w;
#pragma unroll
  for (int m = 1; m < 32; m <<= 1) {
    s += __shfl_xor(s, m, 32);
    sq += __shfl_xor(sq, m, 32);
  }
  __shared__ float ss[8], ssq[8];
  const int w = t >> 5, lane = t & 31;
  if (lane == 0) { ss[w] = s; ssq[w] = sq; }
  __syncthreads();
  float tot = 0.f, totq = 0.f;
#pragma unroll
  for (int i = 0; i < 8; ++i) { tot += ss[i]; totq += ssq[i]; }
  const float mu = tot * (1.0f / EDIM);
  const float var = totq * (1.0f / EDIM) - mu * mu;
  const float inv = rsqrtf(var + 1e-5f);
  float4 g = ((const float4*)gamma)[t];
  float4 bb = ((const float4*)beta)[t];
  _Float16* o = xn + (size_t)row * EDIM + t * 4;
  o[0] = (_Float16)((v.x - mu) * inv * g.x + bb.x);
  o[1] = (_Float16)((v.y - mu) * inv * g.y + bb.y);
  o[2] = (_Float16)((v.z - mu) * inv * g.z + bb.z);
  o[3] = (_Float16)((v.w - mu) * inv * g.w + bb.w);
}

// ---------------------------------------------------------------------------
// 2. Weight convert+transpose: W f32 [K][N] -> WT f16 [N][K]  (K=N=1024)
// ---------------------------------------------------------------------------
__global__ __launch_bounds__(256) void convert_transpose(
    const float* __restrict__ W, _Float16* __restrict__ WT) {
  __shared__ float tile[32][33];
  const int k0 = blockIdx.x * 32, n0 = blockIdx.y * 32;
  const int tx = threadIdx.x & 31, ty = threadIdx.x >> 5;  // ty: 0..7
#pragma unroll
  for (int i = 0; i < 32; i += 8)
    tile[ty + i][tx] = W[(size_t)(k0 + ty + i) * EDIM + n0 + tx];
  __syncthreads();
#pragma unroll
  for (int i = 0; i < 32; i += 8)
    WT[(size_t)(n0 + ty + i) * EDIM + k0 + tx] = (_Float16)tile[tx][ty + i];
}

// ---------------------------------------------------------------------------
// 3/5. GEMM: C[M=8192, N=1024] = A(f16) @ BT^T(f16) (+bias)(*scale)
//      FINAL=false: store f16;  FINAL=true: += residual, store f32
//      Block tile 128x128, 8 waves (4x2) of 32x64, K-step 32.
//      Double-buffered LDS staged by async-to-LDS; one barrier per K step.
// ---------------------------------------------------------------------------
#define TILEH (128 * 32)  // halfs per LDS tile buffer

template <bool FINAL>
__global__ __launch_bounds__(256) void gemm128(
    const _Float16* __restrict__ A, const _Float16* __restrict__ BT,
    const float* __restrict__ bias, float scale,
    const float* __restrict__ residual, _Float16* __restrict__ outH,
    float* __restrict__ outF) {
  __shared__ __attribute__((aligned(16))) _Float16 as_[2 * TILEH];
  __shared__ __attribute__((aligned(16))) _Float16 bs_[2 * TILEH];
  const int t = threadIdx.x;
  const int m0 = blockIdx.x * 128, n0 = blockIdx.y * 128;
  const int w = t >> 5;
  const int wm = (w & 3) * 32;   // wave row offset within tile
  const int wn = (w >> 2) * 64;  // wave col offset within tile
  const int lr = t >> 1;         // 0..127
  const int lc = (t & 1) * 16;   // 0 or 16

  v8f acc[2][4];
#pragma unroll
  for (int i = 0; i < 2; ++i)
#pragma unroll
    for (int j = 0; j < 4; ++j) acc[i][j] = v8f_zero();

  const _Float16* gA = A + (size_t)(m0 + lr) * EDIM + lc;
  const _Float16* gB = BT + (size_t)(n0 + lr) * EDIM + lc;

  // stage tile at k0 into LDS buffer `buf` (each thread: 2x b128 per matrix)
  auto stage = [&](int k0, int buf) {
    _Float16* sa = as_ + buf * TILEH + lr * 32 + lc;
    _Float16* sb = bs_ + buf * TILEH + lr * 32 + lc;
#if USE_ASYNC_COPY
    async_b128(gA + k0, sa);
    async_b128(gA + k0 + 8, sa + 8);
    async_b128(gB + k0, sb);
    async_b128(gB + k0 + 8, sb + 8);
#else
    const v8h* pa = (const v8h*)(gA + k0);
    const v8h* pb = (const v8h*)(gB + k0);
    ((v8h*)sa)[0] = pa[0];
    ((v8h*)sa)[1] = pa[1];
    ((v8h*)sb)[0] = pb[0];
    ((v8h*)sb)[1] = pb[1];
    if (k0 + 32 < EDIM) {  // global_prefetch_b8 for the next K tile
      __builtin_prefetch(gA + k0 + 32, 0, 1);
      __builtin_prefetch(gB + k0 + 32, 0, 1);
    }
#endif
  };

  stage(0, 0);
  int cur = 0;
  for (int k0 = 0; k0 < EDIM; k0 += 32) {
    async_wait0();     // my async portions of buf[cur] are complete
    __syncthreads();   // everyone's portions complete; prev compute done
    if (k0 + 32 < EDIM) stage(k0 + 32, cur ^ 1);  // overlap with WMMA below

    const _Float16* ab = as_ + cur * TILEH;
    const _Float16* bb = bs_ + cur * TILEH;
    v16h af[2], bf[4];
#pragma unroll
    for (int i = 0; i < 2; ++i) af[i] = load_frag_a(ab, wm + i * 16, 0, 32);
#pragma unroll
    for (int j = 0; j < 4; ++j) bf[j] = load_frag_b(bb, wn + j * 16, 0, 32);
#pragma unroll
    for (int i = 0; i < 2; ++i)
#pragma unroll
      for (int j = 0; j < 4; ++j) acc[i][j] = wmma_f16(af[i], bf[j], acc[i][j]);
    cur ^= 1;
  }

  const int lane = t & 31;
  const int colL = lane & 15;
  const int rsel = (lane >> 4) * 8;
#pragma unroll
  for (int i = 0; i < 2; ++i)
#pragma unroll
    for (int j = 0; j < 4; ++j) {
      const int col = n0 + wn + j * 16 + colL;
      const float bv = bias[col];
#pragma unroll
      for (int e = 0; e < 8; ++e) {
        const int row = m0 + wm + i * 16 + e + rsel;
        const float val = acc[i][j][e] + bv;
        if constexpr (FINAL) {
          outF[(size_t)row * EDIM + col] =
              val + residual[(size_t)row * EDIM + col];
        } else {
          outH[(size_t)row * EDIM + col] = (_Float16)(val * scale);
        }
      }
    }
}

// ---------------------------------------------------------------------------
// 4. Causal flash attention. Grid (S/64, H, B), 128 threads (4 waves).
//    Wave w owns query rows q0+w*16 .. +15. Q pre-scaled by 1/sqrt(D).
// ---------------------------------------------------------------------------
__global__ __launch_bounds__(128) void attn_flash(
    const _Float16* __restrict__ Qh, const _Float16* __restrict__ Kh,
    const _Float16* __restrict__ Vh, _Float16* __restrict__ Oh) {
  __shared__ __attribute__((aligned(16))) _Float16 qs[64 * 64];  // Q, then P
  __shared__ __attribute__((aligned(16))) _Float16 ks[64 * 64];  // K rows
  __shared__ __attribute__((aligned(16))) _Float16 vts[64 * 64]; // V^T [d][j]

  const int t = threadIdx.x;
  const int w = t >> 5, lane = t & 31;
  const int q0 = blockIdx.x * 64;
  const int h = blockIdx.y, b = blockIdx.z;
  const size_t base = (size_t)b * SEQ * HD + (size_t)h * DHEAD;
  const int lr = t >> 1, lc = (t & 1) * 32;  // cooperative tile loads
  const int colL = lane & 15;
  const int rsel = (lane >> 4) * 8;
  const int rowbase = q0 + w * 16 + rsel;  // + e = absolute query row

  {  // load Q tile (64x64)
    const v8h* g = (const v8h*)(Qh + base + (size_t)(q0 + lr) * HD + lc);
    v8h* sd = (v8h*)(qs + lr * 64 + lc);
    sd[0] = g[0]; sd[1] = g[1]; sd[2] = g[2]; sd[3] = g[3];
  }
  __syncthreads();
  v16h qf[2];
  qf[0] = load_frag_a(qs, w * 16, 0, 64);
  qf[1] = load_frag_a(qs, w * 16, 32, 64);

  v8f o[4];
#pragma unroll
  for (int j = 0; j < 4; ++j) o[j] = v8f_zero();
  float m_i[8], l_i[8];
#pragma unroll
  for (int e = 0; e < 8; ++e) { m_i[e] = -3.0e38f; l_i[e] = 0.0f; }

  for (int j0 = 0; j0 <= q0; j0 += 64) {
    __syncthreads();  // previous iteration done with ks/vts; qs free
    {  // K tile rows via async-to-LDS; V transposed synchronously
      const _Float16* gk = Kh + base + (size_t)(j0 + lr) * HD + lc;
      _Float16* sk = ks + lr * 64 + lc;
#if USE_ASYNC_COPY
      async_b128(gk, sk);
      async_b128(gk + 8, sk + 8);
      async_b128(gk + 16, sk + 16);
      async_b128(gk + 24, sk + 24);
#else
      ((v8h*)sk)[0] = ((const v8h*)gk)[0];
      ((v8h*)sk)[1] = ((const v8h*)gk)[1];
      ((v8h*)sk)[2] = ((const v8h*)gk)[2];
      ((v8h*)sk)[3] = ((const v8h*)gk)[3];
#endif
      const _Float16* gv = Vh + base + (size_t)(j0 + lr) * HD + lc;
#pragma unroll
      for (int c = 0; c < 32; ++c) vts[(lc + c) * 64 + lr] = gv[c];
    }
    async_wait0();
    __syncthreads();

    // S = Q @ K^T  (16x64 per wave)
    v8f s[4];
#pragma unroll
    for (int nt = 0; nt < 4; ++nt) {
      s[nt] = v8f_zero();
      s[nt] = wmma_f16(qf[0], load_frag_b(ks, nt * 16, 0, 64), s[nt]);
      s[nt] = wmma_f16(qf[1], load_frag_b(ks, nt * 16, 32, 64), s[nt]);
    }
    if (j0 == q0) {  // diagonal block: causal mask
#pragma unroll
      for (int nt = 0; nt < 4; ++nt)
#pragma unroll
        for (int e = 0; e < 8; ++e)
          if (j0 + nt * 16 + colL > rowbase + e) s[nt][e] = -3.0e38f;
    }

    // online softmax: row = element e + lane-half; reduce across 16 lanes
    float mx[8];
#pragma unroll
    for (int e = 0; e < 8; ++e)
      mx[e] = fmaxf(fmaxf(s[0][e], s[1][e]), fmaxf(s[2][e], s[3][e]));
#pragma unroll
    for (int msk = 1; msk < 16; msk <<= 1)
#pragma unroll
      for (int e = 0; e < 8; ++e) mx[e] = fmaxf(mx[e], __shfl_xor(mx[e], msk, 32));

    float alpha[8], rs[8];
#pragma unroll
    for (int e = 0; e < 8; ++e) {
      const float newm = fmaxf(m_i[e], mx[e]);
      alpha[e] = __expf(m_i[e] - newm);
      m_i[e] = newm;
      rs[e] = 0.0f;
    }
#pragma unroll
    for (int nt = 0; nt < 4; ++nt)
#pragma unroll
      for (int e = 0; e < 8; ++e) {
        const float p = __expf(s[nt][e] - m_i[e]);
        s[nt][e] = p;
        rs[e] += p;
      }
#pragma unroll
    for (int msk = 1; msk < 16; msk <<= 1)
#pragma unroll
      for (int e = 0; e < 8; ++e) rs[e] += __shfl_xor(rs[e], msk, 32);
#pragma unroll
    for (int e = 0; e < 8; ++e) l_i[e] = l_i[e] * alpha[e] + rs[e];
#pragma unroll
    for (int nt = 0; nt < 4; ++nt)
#pragma unroll
      for (int e = 0; e < 8; ++e) o[nt][e] *= alpha[e];

    // P (C layout) -> LDS (A layout source), per-wave region of qs
#pragma unroll
    for (int nt = 0; nt < 4; ++nt)
#pragma unroll
      for (int e = 0; e < 8; ++e)
        qs[(w * 16 + e + rsel) * 64 + nt * 16 + colL] = (_Float16)s[nt][e];
    __syncthreads();

    // O += P @ V
    v16h pf0 = load_frag_a(qs, w * 16, 0, 64);
    v16h pf1 = load_frag_a(qs, w * 16, 32, 64);
#pragma unroll
    for (int nt = 0; nt < 4; ++nt) {
      o[nt] = wmma_f16(pf0, load_frag_b(vts, nt * 16, 0, 64), o[nt]);
      o[nt] = wmma_f16(pf1, load_frag_b(vts, nt * 16, 32, 64), o[nt]);
    }
  }

  // normalize and store
  float inv[8];
#pragma unroll
  for (int e = 0; e < 8; ++e) inv[e] = 1.0f / l_i[e];
#pragma unroll
  for (int nt = 0; nt < 4; ++nt)
#pragma unroll
    for (int e = 0; e < 8; ++e) {
      const int row = rowbase + e;
      Oh[base + (size_t)row * HD + nt * 16 + colL] =
          (_Float16)(o[nt][e] * inv[e]);
    }
}

// ---------------------------------------------------------------------------
extern "C" void kernel_launch(void* const* d_in, const int* in_sizes, int n_in,
                              void* d_out, int out_size, void* d_ws,
                              size_t ws_size, hipStream_t stream) {
  const float* x     = (const float*)d_in[0];
  const float* gamma = (const float*)d_in[1];
  const float* beta  = (const float*)d_in[2];
  const float* Wq    = (const float*)d_in[3];
  const float* bq    = (const float*)d_in[4];
  const float* Wk    = (const float*)d_in[5];
  const float* bk    = (const float*)d_in[6];
  const float* Wv    = (const float*)d_in[7];
  const float* bv    = (const float*)d_in[8];
  const float* Wo    = (const float*)d_in[9];
  const float* bo    = (const float*)d_in[10];
  float* out = (float*)d_out;

  char* ws = (char*)d_ws;
  const size_t MB = 1024ull * 1024ull;
  _Float16* xnH = (_Float16*)(ws + 0 * MB);
  _Float16* WqT = (_Float16*)(ws + 16 * MB);
  _Float16* WkT = (_Float16*)(ws + 18 * MB);
  _Float16* WvT = (_Float16*)(ws + 20 * MB);
  _Float16* WoT = (_Float16*)(ws + 22 * MB);
  _Float16* Qh  = (_Float16*)(ws + 24 * MB);
  _Float16* Kh  = (_Float16*)(ws + 40 * MB);
  _Float16* Vh  = (_Float16*)(ws + 56 * MB);
  _Float16* Ah  = (_Float16*)(ws + 72 * MB);

  layernorm_f16<<<MROWS, 256, 0, stream>>>(x, gamma, beta, xnH);

  dim3 tg(32, 32);
  convert_transpose<<<tg, 256, 0, stream>>>(Wq, WqT);
  convert_transpose<<<tg, 256, 0, stream>>>(Wk, WkT);
  convert_transpose<<<tg, 256, 0, stream>>>(Wv, WvT);
  convert_transpose<<<tg, 256, 0, stream>>>(Wo, WoT);

  dim3 gg(MROWS / 128, EDIM / 128);
  const float qscale = 0.125f;  // 1/sqrt(D)
  gemm128<false><<<gg, 256, 0, stream>>>(xnH, WqT, bq, qscale, nullptr, Qh, nullptr);
  gemm128<false><<<gg, 256, 0, stream>>>(xnH, WkT, bk, 1.0f, nullptr, Kh, nullptr);
  gemm128<false><<<gg, 256, 0, stream>>>(xnH, WvT, bv, 1.0f, nullptr, Vh, nullptr);

  attn_flash<<<dim3(SEQ / 64, HEADS, BATCH), 128, 0, stream>>>(Qh, Kh, Vh, Ah);

  gemm128<true><<<gg, 256, 0, stream>>>(Ah, WoT, bo, 1.0f, x, nullptr, out);
}
